// MoELayer_5970004541626
// MI455X (gfx1250) — matrix operand order repaired and checked by reading, not compiled, not run on hardware
//
#include <hip/hip_runtime.h>
#include <hip/hip_bf16.h>
#include <math.h>

// ---------------------------------------------------------------------------
// MoE (top-2 of 8 experts, C=1024, F=4096, N=16384 tokens) for MI455X/gfx1250.
//
//   * Prologue kernels convert x/W1/W2 to bf16; W1/W2 are pre-swizzled into
//     WMMA B-fragment order (lane-contiguous 32B lines). ~500MB of traffic
//     ~= 20us @ 23.3 TB/s -- negligible vs ~220us of bf16-WMMA compute.
//   * Routed FFN: deterministic per-expert token lists, 32-token tiles,
//     v_wmma_f32_16x16x32_bf16, f32 accumulation held in registers
//     (32x1024 out tile). A/B fragments are direct contiguous 32B global
//     loads (bf16 weight set = 128MB -> resident in the 192MB L2); only the
//     h hand-off goes through LDS (8KB). Double-buffered operand loads.
// ---------------------------------------------------------------------------

#define C_DIM 1024
#define F_DIM 4096
#define N_EXP 8
#define N_TOK 16384
#define TILE_M 32
#define MAX_TILES (N_TOK / TILE_M)
#define W_ELEMS ((size_t)C_DIM * F_DIM)        // per-expert weight elements

typedef __attribute__((ext_vector_type(16))) __bf16 v16bf;
typedef __attribute__((ext_vector_type(8)))  float  v8f;

__device__ __forceinline__ unsigned short f2bf(float f) {
    union { float f; unsigned int u; } v; v.f = f;
    unsigned int u = v.u;
    unsigned int r = (u + 0x7FFFu + ((u >> 16) & 1u)) >> 16;   // RNE
    return (unsigned short)r;
}

__device__ __forceinline__ float gelu_exact(float x) {
    return 0.5f * x * (1.0f + erff(x * 0.70710678118654752f));
}

// ---------------------------------------------------------------------------
// Kernel 0: zero the output (accumulated via atomics; re-zero each call)
// ---------------------------------------------------------------------------
__global__ void moe_zero_out(float* __restrict__ out) {
    int idx = blockIdx.x * blockDim.x + threadIdx.x;
    float4* o4 = (float4*)out;
    if (idx < (N_TOK * C_DIM) / 4) o4[idx] = make_float4(0.f, 0.f, 0.f, 0.f);
}

// ---------------------------------------------------------------------------
// Conversion kernels (prologue; off the critical path)
// ---------------------------------------------------------------------------
__global__ void conv_x_bf16(const float* __restrict__ x,
                            unsigned short* __restrict__ xbf) {
    int tid = blockIdx.x * blockDim.x + threadIdx.x;
    size_t s = (size_t)tid * 4;
    if (s >= (size_t)N_TOK * C_DIM) return;
    float4 v = *(const float4*)(x + s);
    unsigned short tmp[4] = { f2bf(v.x), f2bf(v.y), f2bf(v.z), f2bf(v.w) };
    *(uint2*)(xbf + s) = *(const uint2*)tmp;
}

// W1 (E,C,F) row-major -> bf16 fragment order:
//   dst[(((e*256 + ft)*32 + kc)*512) + (j + 16*(kk>>4))*16 + (kk&15)] = W1[e][kc*32+kk][ft*16+j]
__global__ void conv_w1_bf16(const float* __restrict__ W1,
                             unsigned short* __restrict__ W1bf) {
    size_t tid = (size_t)blockIdx.x * blockDim.x + threadIdx.x;
    size_t s = tid * 4;
    if (s >= (size_t)N_EXP * W_ELEMS) return;
    int f = (int)(s & 4095);
    int k = (int)((s >> 12) & 1023);
    int e = (int)(s >> 22);
    float4 v = *(const float4*)(W1 + s);
    int ft = f >> 4, j = f & 15;
    int kc = k >> 5, kk = k & 31;
    size_t d = (((size_t)(e * 256 + ft) * 32 + kc) * 512) + (j + 16 * (kk >> 4)) * 16 + (kk & 15);
    W1bf[d] = f2bf(v.x); W1bf[d + 16] = f2bf(v.y);
    W1bf[d + 32] = f2bf(v.z); W1bf[d + 48] = f2bf(v.w);
}

// W2 (E,F,C) row-major -> bf16 fragment order:
//   dst[(((e*64 + ct)*128 + kt)*512) + (j + 16*(kk>>4))*16 + (kk&15)] = W2[e][kt*32+kk][ct*16+j]
__global__ void conv_w2_bf16(const float* __restrict__ W2,
                             unsigned short* __restrict__ W2bf) {
    size_t tid = (size_t)blockIdx.x * blockDim.x + threadIdx.x;
    size_t s = tid * 4;
    if (s >= (size_t)N_EXP * W_ELEMS) return;
    int c = (int)(s & 1023);
    int f = (int)((s >> 10) & 4095);
    int e = (int)(s >> 22);
    float4 v = *(const float4*)(W2 + s);
    int ct = c >> 4, j = c & 15;
    int kt = f >> 5, kk = f & 31;
    size_t d = (((size_t)(e * 64 + ct) * 128 + kt) * 512) + (j + 16 * (kk >> 4)) * 16 + (kk & 15);
    W2bf[d] = f2bf(v.x); W2bf[d + 16] = f2bf(v.y);
    W2bf[d + 32] = f2bf(v.z); W2bf[d + 48] = f2bf(v.w);
}

// ---------------------------------------------------------------------------
// Kernel 1: gate. One wave per token -> 8 logits, top-2, softmax.
// ---------------------------------------------------------------------------
__global__ void moe_gate(const float* __restrict__ x,
                         const float* __restrict__ Wg,
                         int*   __restrict__ e0_arr,
                         int*   __restrict__ e1_arr,
                         float* __restrict__ w0_arr,
                         float* __restrict__ w1_arr) {
    const int wave = threadIdx.x >> 5;
    const int lane = threadIdx.x & 31;
    const int n = blockIdx.x * 8 + wave;
    if (n >= N_TOK) return;

    float s[N_EXP];
#pragma unroll
    for (int e = 0; e < N_EXP; ++e) s[e] = 0.f;
    const float* xrow = x + (size_t)n * C_DIM;
#pragma unroll 4
    for (int i = 0; i < C_DIM / 32; ++i) {
        int c = lane + 32 * i;
        float xv = xrow[c];
        const float4* wp = (const float4*)(Wg + c * N_EXP);
        float4 g0 = wp[0], g1 = wp[1];
        s[0] += xv * g0.x; s[1] += xv * g0.y; s[2] += xv * g0.z; s[3] += xv * g0.w;
        s[4] += xv * g1.x; s[5] += xv * g1.y; s[6] += xv * g1.z; s[7] += xv * g1.w;
    }
#pragma unroll
    for (int off = 16; off > 0; off >>= 1)
#pragma unroll
        for (int e = 0; e < N_EXP; ++e) s[e] += __shfl_xor(s[e], off, 32);
    if (lane == 0) {
        int i0 = 0;
#pragma unroll
        for (int e = 1; e < N_EXP; ++e) if (s[e] > s[i0]) i0 = e;
        int i1 = (i0 == 0) ? 1 : 0;
#pragma unroll
        for (int e = 0; e < N_EXP; ++e) if (e != i0 && s[e] > s[i1]) i1 = e;
        float z  = __expf(s[i1] - s[i0]);
        float sm = 1.0f + z;
        e0_arr[n] = i0; e1_arr[n] = i1;
        w0_arr[n] = 1.0f / sm;
        w1_arr[n] = z / sm;
    }
}

// ---------------------------------------------------------------------------
// Kernel 2: deterministic per-expert compaction (token-order, no atomics).
// ---------------------------------------------------------------------------
__global__ void moe_compact(const int* __restrict__ e0_arr,
                            const int* __restrict__ e1_arr,
                            const float* __restrict__ w0_arr,
                            const float* __restrict__ w1_arr,
                            int*   __restrict__ tok_list,
                            float* __restrict__ gate_list,
                            int*   __restrict__ counts) {
    __shared__ int scan[256];
    const int e = blockIdx.x;
    const int t = threadIdx.x;
    int base = 0;
    for (int chunk = 0; chunk < N_TOK; chunk += 256) {
        int n = chunk + t;
        int m0 = (e0_arr[n] == e);
        int m1 = (e1_arr[n] == e);
        int flag = m0 | m1;
        float w = m0 ? w0_arr[n] : w1_arr[n];
        scan[t] = flag;
        __syncthreads();
        for (int off = 1; off < 256; off <<= 1) {
            int v = scan[t];
            int a = (t >= off) ? scan[t - off] : 0;
            __syncthreads();
            scan[t] = v + a;
            __syncthreads();
        }
        int incl = scan[t];
        int total = scan[255];
        if (flag) {
            int pos = base + incl - 1;
            tok_list[e * N_TOK + pos]  = n;
            gate_list[e * N_TOK + pos] = w;
        }
        base += total;
        __syncthreads();
    }
    if (t == 0) counts[e] = base;
}

// ---------------------------------------------------------------------------
// Kernel 3: routed FFN. grid = (MAX_TILES, E), block = 256 (8 waves).
// ---------------------------------------------------------------------------
__global__ void __launch_bounds__(256)
moe_ffn(const unsigned short* __restrict__ xbf,
        const unsigned short* __restrict__ W1bf,
        const unsigned short* __restrict__ W2bf,
        const int*   __restrict__ tok_list,
        const float* __restrict__ gate_list,
        const int*   __restrict__ counts,
        float* __restrict__ out) {
    const int e = blockIdx.y;
    const int cnt = counts[e];
    const int start = blockIdx.x * TILE_M;
    if (start >= cnt) return;

    __shared__ unsigned short ldsH[8 * 512];   // h pass (32x128 bf16, frag order)
    __shared__ int   s_tok[TILE_M];
    __shared__ float s_cw[TILE_M];

    const int t    = threadIdx.x;
    const int wave = t >> 5;
    const int lane = t & 31;

    if (t < TILE_M) {
        int idx = start + t;
        if (idx < cnt) {
            s_tok[t] = tok_list[e * N_TOK + idx];
            s_cw[t]  = gate_list[e * N_TOK + idx];
        } else {
            s_tok[t] = 0;
            s_cw[t]  = 0.0f;      // padded rows contribute exactly 0
        }
    }
    __syncthreads();

    // A-fragment base pointers: lane L covers row m=L%16 (+16 for tile 1),
    // K-halves selected by L/16 -> one contiguous 32B read per fragment.
    const unsigned short* arow0 =
        xbf + (size_t)s_tok[lane & 15] * C_DIM + 16 * (lane >> 4);
    const unsigned short* arow1 =
        xbf + (size_t)s_tok[16 + (lane & 15)] * C_DIM + 16 * (lane >> 4);

    const unsigned short* W1e = W1bf + (size_t)e * W_ELEMS;
    const unsigned short* W2e = W2bf + (size_t)e * W_ELEMS;
    // Phase-B base: offsets from here are q*1MB (col group) + kt*1KB (K chunk)
    const unsigned short* w2base = W2e + (size_t)wave * 128 * 512 + lane * 16;

    v8f accO[16];
#pragma unroll
    for (int i = 0; i < 16; ++i)
#pragma unroll
        for (int r = 0; r < 8; ++r) accO[i][r] = 0.0f;

    for (int pass = 0; pass < F_DIM / 128; ++pass) {
        // ------------- Phase A: h = gelu(x @ W1[:, pass*128 ..]) -------------
        v8f h0, h1;
#pragma unroll
        for (int r = 0; r < 8; ++r) { h0[r] = 0.0f; h1[r] = 0.0f; }

        const unsigned short* bptr =
            W1e + (size_t)(pass * 8 + wave) * 32 * 512 + lane * 16;
        // software pipeline: keep next iteration's operands in flight
        v16bf a0c = *(const v16bf*)(arow0);
        v16bf a1c = *(const v16bf*)(arow1);
        v16bf bc  = *(const v16bf*)(bptr);
        for (int kc = 0; kc < C_DIM / 32 - 1; ++kc) {
            v16bf a0n = *(const v16bf*)(arow0 + (kc + 1) * 32);
            v16bf a1n = *(const v16bf*)(arow1 + (kc + 1) * 32);
            v16bf bn  = *(const v16bf*)(bptr + 512);
            __builtin_prefetch(bptr + 1024, 0, 3);
            h0 = __builtin_amdgcn_wmma_f32_16x16x32_bf16(
                false, a0c, false, bc, (short)0, h0, false, false);
            h1 = __builtin_amdgcn_wmma_f32_16x16x32_bf16(
                false, a1c, false, bc, (short)0, h1, false, false);
            a0c = a0n; a1c = a1n; bc = bn;
            bptr += 512;
        }
        h0 = __builtin_amdgcn_wmma_f32_16x16x32_bf16(
            false, a0c, false, bc, (short)0, h0, false, false);
        h1 = __builtin_amdgcn_wmma_f32_16x16x32_bf16(
            false, a1c, false, bc, (short)0, h1, false, false);

        __syncthreads();   // previous Phase B done reading ldsH
        // gelu + gate scale -> bf16 -> ldsH in A-frag order
#pragma unroll
        for (int rt = 0; rt < 2; ++rt) {
#pragma unroll
            for (int r = 0; r < 8; ++r) {
                int m = r + 8 * (lane >> 4) + rt * 16;
                float v = gelu_exact(rt ? h1[r] : h0[r]) * s_cw[m];
                int fl = wave * 16 + (lane & 15);       // f within the pass
                int kt2 = fl >> 5, k = fl & 31;
                int lp = (m & 15) + 16 * (k >> 4);
                int jp = k & 15;
                ldsH[(kt2 * 2 + rt) * 512 + lp * 16 + jp] = f2bf(v);
            }
        }
        __syncthreads();

        // ------------- Phase B: out += h @ W2[pass*128 .., :] ---------------
#pragma unroll
        for (int kt2 = 0; kt2 < 4; ++kt2) {
            v16bf a0 = *(const v16bf*)&ldsH[(kt2 * 2 + 0) * 512 + lane * 16];
            v16bf a1 = *(const v16bf*)&ldsH[(kt2 * 2 + 1) * 512 + lane * 16];
            const size_t ktoff = (size_t)(pass * 4 + kt2) * 512;
            // one-step rotate so the next B frag is in flight during WMMA
            v16bf b2c = *(const v16bf*)(w2base + ktoff);
#pragma unroll
            for (int q = 0; q < 8; ++q) {
                v16bf b2n;
                if (q < 7) {
                    b2n = *(const v16bf*)(w2base + (size_t)(q + 1) * 524288 + ktoff);
                    __builtin_prefetch(w2base + (size_t)(q + 1) * 524288 + ktoff + 512,
                                       0, 3);
                }
                accO[2 * q + 0] = __builtin_amdgcn_wmma_f32_16x16x32_bf16(
                    false, a0, false, b2c, (short)0, accO[2 * q + 0], false, false);
                accO[2 * q + 1] = __builtin_amdgcn_wmma_f32_16x16x32_bf16(
                    false, a1, false, b2c, (short)0, accO[2 * q + 1], false, false);
                if (q < 7) b2c = b2n;
            }
        }
    }

    // Scatter-add: one row pointer per (rt,r); 8 atomics at immediate offsets.
    // Each token row receives exactly two expert contributions total; f32 add
    // of two values over a zeroed base is order-independent bitwise.
#pragma unroll
    for (int rt = 0; rt < 2; ++rt) {
#pragma unroll
        for (int r = 0; r < 8; ++r) {
            int m = r + 8 * (lane >> 4) + rt * 16;
            if (start + m < cnt) {
                float* rowp = out + (size_t)s_tok[m] * C_DIM
                                  + wave * 16 + (lane & 15);
#pragma unroll
                for (int q = 0; q < 8; ++q)
                    atomicAdd(rowp + q * 128, accO[2 * q + rt][r]);
            }
        }
    }
}

// ---------------------------------------------------------------------------
// Launch
// ---------------------------------------------------------------------------
extern "C" void kernel_launch(void* const* d_in, const int* in_sizes, int n_in,
                              void* d_out, int out_size, void* d_ws, size_t ws_size,
                              hipStream_t stream) {
    const float* x  = (const float*)d_in[0];
    const float* Wg = (const float*)d_in[1];
    const float* W1 = (const float*)d_in[2];
    const float* W2 = (const float*)d_in[3];
    float* out = (float*)d_out;

    // workspace layout
    char* ws = (char*)d_ws;
    int*   e0_arr   = (int*)ws;      ws += N_TOK * sizeof(int);
    int*   e1_arr   = (int*)ws;      ws += N_TOK * sizeof(int);
    float* w0_arr   = (float*)ws;    ws += N_TOK * sizeof(float);
    float* w1_arr   = (float*)ws;    ws += N_TOK * sizeof(float);
    int*   counts   = (int*)ws;      ws += 256;
    int*   tok_list = (int*)ws;      ws += (size_t)N_EXP * N_TOK * sizeof(int);
    float* gate_lst = (float*)ws;    ws += (size_t)N_EXP * N_TOK * sizeof(float);
    unsigned short* xbf  = (unsigned short*)ws;  ws += (size_t)N_TOK * C_DIM * 2;
    unsigned short* W1bf = (unsigned short*)ws;  ws += (size_t)N_EXP * W_ELEMS * 2;
    unsigned short* W2bf = (unsigned short*)ws;  ws += (size_t)N_EXP * W_ELEMS * 2;

    // 0) zero output (replay-safe)
    moe_zero_out<<<(N_TOK * C_DIM / 4 + 255) / 256, 256, 0, stream>>>(out);
    // prologue: bf16 conversion + fragment swizzle (tiny vs compute)
    conv_x_bf16<<<(N_TOK * C_DIM / 4 + 255) / 256, 256, 0, stream>>>(x, xbf);
    {
        int blocks = (int)((N_EXP * W_ELEMS / 4 + 255) / 256);
        conv_w1_bf16<<<blocks, 256, 0, stream>>>(W1, W1bf);
        conv_w2_bf16<<<blocks, 256, 0, stream>>>(W2, W2bf);
    }
    // 1) gate
    moe_gate<<<N_TOK / 8, 256, 0, stream>>>(x, Wg, e0_arr, e1_arr, w0_arr, w1_arr);
    // 2) deterministic compaction
    moe_compact<<<N_EXP, 256, 0, stream>>>(e0_arr, e1_arr, w0_arr, w1_arr,
                                           tok_list, gate_lst, counts);
    // 3) routed bf16-WMMA FFN
    dim3 grid(MAX_TILES, N_EXP);
    moe_ffn<<<grid, 256, 0, stream>>>(xbf, W1bf, W2bf, tok_list, gate_lst,
                                      counts, out);
}
